// LSTM_Encoder_70497593197122
// MI455X (gfx1250) — matrix operand order repaired and checked by reading, not compiled, not run on hardware
//
#include <hip/hip_runtime.h>
#include <hip/hip_bf16.h>
#include <stdint.h>

// ---------------------------------------------------------------------------
// LSTM encoder, B=2048, T=256, H=512, IN=1, CDNA5 (gfx1250, wave32, WMMA).
//
// Strategy:
//  * Batch rows are independent through the recurrence -> each workgroup owns
//    a 16-row batch tile and runs all 256 timesteps with h/c resident in LDS.
//  * Recurrent GEMM (16 x 2048 x 512) done with v_wmma_f32_16x16x32_bf16.
//    Weights pre-converted to bf16 and pre-swizzled into the per-lane WMMA
//    B-operand layout; gate columns interleaved as n' = 4*j + gate so every
//    16-wide N-tile holds i,f,g,o for 4 hidden units (local elementwise).
//  * bf16 h is double-buffered in LDS (A operand); c stays fp32 in LDS.
//  * Rank-1 x/bias terms + activations + final 512->2 projection in VALU,
//    with an LDS tree reduction across the 8 waves for the projection.
// ---------------------------------------------------------------------------

#define BT 2048
#define TT 256
#define HH 512
#define GG 2048   // 4*H

typedef float  v8f   __attribute__((ext_vector_type(8)));
typedef __bf16 v16bf __attribute__((ext_vector_type(16)));

union ABPack {
    uint4 q[2];
    v16bf v;
};

__device__ __forceinline__ float fast_sigmoid(float x) {
    return __builtin_amdgcn_rcpf(1.0f + __expf(-x));
}
__device__ __forceinline__ float fast_tanh(float x) {
    // tanh(x) = 1 - 2/(exp(2x)+1)
    return 1.0f - 2.0f * __builtin_amdgcn_rcpf(__expf(2.0f * x) + 1.0f);
}

// ---------------------------------------------------------------------------
// Prep kernel 1: W_hh (2048x512 f32, gate-major rows) -> bf16, gate-interleaved
// column order n' = 4*j + g, packed into WMMA B-operand tiles:
//   tile (nt 0..127, kt 0..15), per tile: lane = (kk>>4)*16 + (n'&15),
//   16 contiguous K values per lane (32B) -> two global_load_b128 at use time.
// ---------------------------------------------------------------------------
__global__ __launch_bounds__(256) void prep_weights(const float* __restrict__ whh,
                                                    __bf16* __restrict__ wsw) {
    int idx = blockIdx.x * 256 + threadIdx.x;     // 0 .. 2048*512-1
    int np  = idx >> 9;                           // swizzled column n'
    int k   = idx & 511;                          // K (source hidden index)
    int g = np & 3, j = np >> 2;
    int orig = g * 512 + j;                       // original gate row
    float v = whh[(size_t)orig * 512 + k];
    int nt = np >> 4, N = np & 15;
    int kt = k >> 5, kk = k & 31;
    int half = kk >> 4, k16 = kk & 15;
    int lane = half * 16 + N;
    wsw[(size_t)((nt * 16 + kt) * 512) + lane * 16 + k16] = (__bf16)v;
}

// ---------------------------------------------------------------------------
// Prep kernel 2: fused bias + x-path weights, gate-interleaved.
// ---------------------------------------------------------------------------
__global__ __launch_bounds__(256) void prep_small(const float* __restrict__ wih,
                                                  const float* __restrict__ bih,
                                                  const float* __restrict__ bhh,
                                                  float* __restrict__ bias_sum,
                                                  float* __restrict__ wih0,
                                                  float* __restrict__ wih1) {
    int np = blockIdx.x * 256 + threadIdx.x;
    if (np >= GG) return;
    int g = np & 3, j = np >> 2;
    int orig = g * 512 + j;
    bias_sum[np] = bih[orig] + bhh[orig];
    wih0[np] = wih[orig * 2 + 0];   // weight_ih[:,0] (x_in column)
    wih1[np] = wih[orig * 2 + 1];   // weight_ih[:,1] (latent column)
}

// ---------------------------------------------------------------------------
// Main kernel: 128 blocks x 256 threads (8 waves), 16 batch rows per block.
// ---------------------------------------------------------------------------
__global__ __launch_bounds__(256, 1) void lstm_main(
    const float* __restrict__ input,   // (B,1)
    const float* __restrict__ lat,     // (B,T)
    const float* __restrict__ h0,      // (B,H)
    const float* __restrict__ c0,      // (B,H)
    const float* __restrict__ eps,     // (B,T)
    const float* __restrict__ wfinal,  // (2,H)
    const float* __restrict__ bfinal,  // (2,)
    const __bf16* __restrict__ wsw,    // swizzled bf16 W_hh
    const float* __restrict__ bias_sum,
    const float* __restrict__ wih0,
    const float* __restrict__ wih1,
    float* __restrict__ out_mu,
    float* __restrict__ out_lv,
    float* __restrict__ out_sp) {

    // Padded strides: 520 bf16 (1040B) keeps 16B alignment and spreads the
    // 16-row A-operand broadcast across all 64 LDS banks.
    __shared__ __align__(16) __bf16 hbuf[2][16][520];  // double-buffered h (bf16)
    __shared__ __align__(16) float  cbuf[16][516];     // cell state (fp32)
    __shared__ __align__(16) float  gscr[8][16][20];   // per-wave 16x16 gate tile
    __shared__ float redm[16][16];                     // projection partials (mean)
    __shared__ float redl[16][16];                     // projection partials (logvar)
    __shared__ float xin[16];                          // fed-back sample
    __shared__ float xlat[16];                         // latent for current t

    const int tid  = threadIdx.x;
    const int wave = tid >> 5;
    const int lane = tid & 31;
    const int bg   = blockIdx.x * 16;

    // ---- init: load h0 (as bf16) and c0 into LDS --------------------------
    for (int idx = tid; idx < 16 * HH; idx += 256) {
        int M = idx >> 9, j = idx & 511;
        hbuf[0][M][j] = (__bf16)h0[(size_t)(bg + M) * HH + j];
        cbuf[M][j]    = c0[(size_t)(bg + M) * HH + j];
    }
    if (tid < 16) {
        xin[tid]  = input[bg + tid];
        xlat[tid] = lat[(size_t)(bg + tid) * TT + 0];
    }
    __syncthreads();

    const int M       = lane & 15;   // batch row within the tile (A/C row)
    const int halfsel = lane >> 4;   // lane half (K/M half selector)

    for (int t = 0; t < TT; ++t) {
        const int cur = t & 1, nxt = cur ^ 1;
        float pm = 0.0f, pl = 0.0f;
        const float xv = xin[M];
        const float lv = xlat[M];

        // ---- phase A: 16 N-tiles of WMMA + local elementwise update -------
        for (int nt = 0; nt < 16; ++nt) {
            const int NT = wave * 16 + nt;           // global 16-col N tile
            v8f acc = {0.f, 0.f, 0.f, 0.f, 0.f, 0.f, 0.f, 0.f};
            const uint4* wq = (const uint4*)wsw + (size_t)NT * 16 * 64 + lane * 2;

            #pragma unroll
            for (int kt = 0; kt < 16; ++kt) {
                ABPack A, Bm;
                const int kb = kt * 32 + halfsel * 16; // A layout: K 0..15 contiguous (+16 for upper half)
                A.q[0] = *(const uint4*)&hbuf[cur][M][kb];
                A.q[1] = *(const uint4*)&hbuf[cur][M][kb + 8];
                Bm.q[0] = wq[0];
                Bm.q[1] = wq[1];
                wq += 64;                             // next 1KB K-tile
                acc = __builtin_amdgcn_wmma_f32_16x16x32_bf16(
                    false, A.v, false, Bm.v, (short)0, acc, false, false);
            }

            // scatter 16x16 f32 gate tile to per-wave LDS scratch
            // (C layout: vgpr r, lane l -> row M = r + 8*(l>>4), col N = l&15)
            #pragma unroll
            for (int r = 0; r < 8; ++r)
                gscr[wave][r + 8 * halfsel][lane & 15] = acc[r];

            // elementwise: each lane handles 2 (row, hidden-unit) pairs
            #pragma unroll
            for (int pp = 0; pp < 2; ++pp) {
                const int dj = (pp << 1) | halfsel;          // 0..3
                const int np = wave * 256 + nt * 16 + dj * 4; // gate-interleaved col
                float4 gv = *(const float4*)&gscr[wave][M][dj * 4];
                float4 bs = *(const float4*)&bias_sum[np];
                float4 w0 = *(const float4*)&wih0[np];
                float4 w1 = *(const float4*)&wih1[np];
                float gi = gv.x + bs.x + xv * w0.x + lv * w1.x;
                float gf = gv.y + bs.y + xv * w0.y + lv * w1.y;
                float gg = gv.z + bs.z + xv * w0.z + lv * w1.z;
                float go = gv.w + bs.w + xv * w0.w + lv * w1.w;
                float si = fast_sigmoid(gi);
                float sf = fast_sigmoid(gf);
                float tg = fast_tanh(gg);
                float so = fast_sigmoid(go);
                const int j = wave * 64 + nt * 4 + dj;
                float cy = sf * cbuf[M][j] + si * tg;
                float hy = so * fast_tanh(cy);
                cbuf[M][j] = cy;
                hbuf[nxt][M][j] = (__bf16)hy;
                pm += hy * wfinal[j];
                pl += hy * wfinal[HH + j];
            }
        }
        redm[M][wave * 2 + halfsel] = pm;
        redl[M][wave * 2 + halfsel] = pl;
        __syncthreads();

        // ---- phase B: reduce projection, sample, write outputs, feed back --
        if (tid < 16) {
            float sm = 0.0f, sl = 0.0f;
            #pragma unroll
            for (int s = 0; s < 16; ++s) { sm += redm[tid][s]; sl += redl[tid][s]; }
            float mean = sm + bfinal[0];
            float logv = sl + bfinal[1];
            size_t ob = (size_t)(bg + tid) * TT + t;
            float samp = mean + eps[ob] * __expf(0.5f * logv);
            out_mu[ob] = mean;
            out_lv[ob] = logv;
            out_sp[ob] = samp;
            xin[tid] = samp;
            if (t + 1 < TT) xlat[tid] = lat[ob + 1];
        }
        __syncthreads();
    }
}

// ---------------------------------------------------------------------------
extern "C" void kernel_launch(void* const* d_in, const int* in_sizes, int n_in,
                              void* d_out, int out_size, void* d_ws, size_t ws_size,
                              hipStream_t stream) {
    const float* input = (const float*)d_in[0];
    const float* lat   = (const float*)d_in[1];
    const float* h0    = (const float*)d_in[2];
    const float* c0    = (const float*)d_in[3];
    const float* eps   = (const float*)d_in[4];
    const float* wih   = (const float*)d_in[5];
    const float* whh   = (const float*)d_in[6];
    const float* bih   = (const float*)d_in[7];
    const float* bhh   = (const float*)d_in[8];
    const float* wf    = (const float*)d_in[9];
    const float* bf    = (const float*)d_in[10];

    // workspace layout: [ 2MB swizzled bf16 W_hh | bias_sum | wih0 | wih1 ]
    char* ws = (char*)d_ws;
    __bf16* wsw      = (__bf16*)ws;
    float*  bias_sum = (float*)(ws + (size_t)GG * HH * sizeof(__bf16));
    float*  w0       = bias_sum + GG;
    float*  w1       = w0 + GG;

    float* out    = (float*)d_out;
    float* out_mu = out;
    float* out_lv = out + (size_t)BT * TT;
    float* out_sp = out + 2 * (size_t)BT * TT;

    prep_weights<<<dim3((GG * HH) / 256), dim3(256), 0, stream>>>(whh, wsw);
    prep_small<<<dim3(GG / 256), dim3(256), 0, stream>>>(wih, bih, bhh, bias_sum, w0, w1);
    lstm_main<<<dim3(BT / 16), dim3(256), 0, stream>>>(
        input, lat, h0, c0, eps, wf, bf, wsw, bias_sum, w0, w1,
        out_mu, out_lv, out_sp);
}
